// GAE_27092653703844
// MI455X (gfx1250) — matrix operand order repaired and checked by reading, not compiled, not run on hardware
//
#include <hip/hip_runtime.h>
#include <hip/hip_bf16.h>

typedef __attribute__((ext_vector_type(16))) __bf16 v16bf;
typedef __attribute__((ext_vector_type(8)))  float  v8f;
typedef __attribute__((ext_vector_type(4)))  unsigned int u32x4;
typedef __attribute__((ext_vector_type(8)))  unsigned int u32x8;

// problem constants (match reference)
#define NCLASS 5
#define DIM    64
#define NU     6144
#define NV     6144
#define NB     4096

// ---------------------------------------------------------------------------
// out[n][e] = sum_c bias[c][e]   (accumulator init with summed bias)
// ---------------------------------------------------------------------------
__global__ void bias_init_kernel(const float* __restrict__ bias,
                                 float* __restrict__ out, int total) {
  int idx = blockIdx.x * blockDim.x + threadIdx.x;
  if (idx >= total) return;
  int e = idx & (DIM - 1);
  float s = 0.f;
#pragma unroll
  for (int c = 0; c < NCLASS; ++c) s += bias[c * DIM + e];
  out[idx] = s;
}

__global__ void relu_kernel(float* __restrict__ x, int n) {
  int i = blockIdx.x * blockDim.x + threadIdx.x;
  if (i < n) x[i] = fmaxf(x[i], 0.f);
}

// ---------------------------------------------------------------------------
// msgT: out[c][e][n] = sum_d src[n][d] * W[c][d][e], stored transposed as
// bf16 hi/lo pair so the SpMM B-fragment is a contiguous 32-byte K-chunk.
// grid = (N/64, 64, C), block = 64
// ---------------------------------------------------------------------------
__global__ void msgT_kernel(const float* __restrict__ src,
                            const float* __restrict__ W,
                            __bf16* __restrict__ out_hi,
                            __bf16* __restrict__ out_lo, int N) {
  int n = blockIdx.x * 64 + threadIdx.x;
  int e = blockIdx.y;
  int c = blockIdx.z;
  const float* srow = src + (size_t)n * DIM;
  const float* wcol = W + (size_t)c * DIM * DIM + e;
  float acc = 0.f;
#pragma unroll
  for (int d = 0; d < DIM; ++d) acc += srow[d] * wcol[(size_t)d * DIM];
  __bf16 hi = (__bf16)acc;
  __bf16 lo = (__bf16)(acc - (float)hi);
  size_t o = (size_t)(c * DIM + e) * N + n;
  out_hi[o] = hi;
  out_lo[o] = lo;
}

// ---------------------------------------------------------------------------
// TDM: issue a 2D tensor_load_to_lds.  D# built in SGPRs:
//   group0: count=1 | lds_addr | global_addr[56:0] | type=2
//   group1: data_size=4B, pad ctrl, huge tensor dims (tiles always in-bounds),
//           tile_dim0 x tile_dim1, tensor_dim0_stride
// Inline asm (not the clang builtin) so it is portable across toolchains.
// ---------------------------------------------------------------------------
__device__ __forceinline__ void tdm_load_2d(unsigned lds_addr, const float* gaddr,
                                            unsigned flags, unsigned tile0,
                                            unsigned tile1, unsigned stride0) {
  unsigned long long ga = (unsigned long long)(size_t)gaddr;
  u32x4 g0;
  g0[0] = 1u;                                             // count=1, user mode
  g0[1] = lds_addr;                                       // LDS byte address
  g0[2] = (unsigned)ga;                                   // global_addr[31:0]
  g0[3] = (unsigned)((ga >> 32) & 0x01ffffffu) | (2u << 30); // [56:32] | type=2
  u32x8 g1;
  g1[0] = flags;                   // data_size=2 (4B) | pad_enable | pad_interval
  g1[1] = 0xffffu << 16;           // tensor_dim0[15:0]  (dim = 0x7fffffff)
  g1[2] = 0x7fffu | (0xffffu << 16); // tensor_dim0[31:16] | tensor_dim1[15:0]
  g1[3] = 0x7fffu | (tile0 << 16); // tensor_dim1[31:16] | tile_dim0
  g1[4] = tile1;                   // tile_dim1 (tile_dim2 = 0)
  g1[5] = stride0;                 // tensor_dim0_stride[31:0]
  g1[6] = 0u;
  g1[7] = 0u;
  asm volatile("tensor_load_to_lds %0, %1" :: "s"(g0), "s"(g1) : "memory");
}

// ---------------------------------------------------------------------------
// Class-summed adjacency contraction:
//   TRANSPOSED=false: out[m][e] += sum_k adj[c][m][k] * msg[c][k][e]
//   TRANSPOSED=true : out[m][e] += sum_k adj[c][k][m] * msg[c][k][e]
// block = 128 threads (4 waves); wave w owns output columns [16w,16w+16) and
// DMAs its own quarter of the 16x128 (or 128x16) f32 adj slab into LDS with
// tensor_load_to_lds (double buffered, TENSORcnt-paced).  TDM line padding
// (1 DWORD per line) gives conflict-free LDS strides of 129 / 17 words.
// adj (0/1 valued) is exact in bf16; msg uses a bf16 hi+lo split so the WMMA
// pair recovers ~16 mantissa bits of the f32 product.
// ---------------------------------------------------------------------------
template <bool TRANSPOSED>
__global__ void __launch_bounds__(128)
spmm_wmma_kernel(const float* __restrict__ adj,       // [C, NU, NV]
                 const __bf16* __restrict__ bT_hi,    // [C, 64, K]
                 const __bf16* __restrict__ bT_lo,    // [C, 64, K]
                 float* __restrict__ out) {           // [M, 64]
  constexpr int Kdim = TRANSPOSED ? NU : NV;
  constexpr int SLAB = TRANSPOSED ? 128 * 17 : 16 * 129;  // f32 words, padded
  __shared__ float adj_s[2][SLAB];

  const int tid  = threadIdx.x;
  const int lane = tid & 31;
  const int wv   = tid >> 5;          // 0..3 : 16-wide e block / TDM quarter
  const int wvu  = __builtin_amdgcn_readfirstlane(wv);
  const int ln16 = lane & 15;
  const int half = lane >> 4;

  const int c     = blockIdx.y;
  const int mbase = blockIdx.x * 16;
  const float* adjc = adj + (size_t)c * NU * NV;

  const int e = wv * 16 + ln16;
  const __bf16* bh_row = bT_hi + (size_t)(c * DIM + e) * Kdim;
  const __bf16* bl_row = bT_lo + (size_t)(c * DIM + e) * Kdim;

  // per-wave quarter-tile TDM setup
  unsigned lds0 = (unsigned)(size_t)&adj_s[0][0];
  unsigned lds1 = (unsigned)(size_t)&adj_s[1][0];
  unsigned lds_off, flags, tile0, tile1;
  const float* gptr;       // this wave's quarter at kb = 0
  size_t gstep;            // global advance per 128-wide K slab
  if (!TRANSPOSED) {
    lds_off = (unsigned)(wvu * 4 * 129 * 4);              // 4 rows of 129 words
    flags   = (2u << 16) | (1u << 20) | (6u << 22);       // 4B, pad 1dw / 128dw
    tile0 = 128; tile1 = 4;
    gptr  = adjc + (size_t)(mbase + wvu * 4) * NV;
    gstep = 128;
  } else {
    lds_off = (unsigned)(wvu * 32 * 17 * 4);              // 32 lines of 17 words
    flags   = (2u << 16) | (1u << 20) | (3u << 22);       // 4B, pad 1dw / 16dw
    tile0 = 16; tile1 = 32;
    gptr  = adjc + (size_t)(wvu * 32) * NV + mbase;
    gstep = (size_t)128 * NV;
  }

  v8f acc = {};

  // prologue: each wave DMAs its quarter of slab 0 into buffer 0
  tdm_load_2d(lds0 + lds_off, gptr, flags, tile0, tile1, NV);
  gptr += gstep;
  int cur = 0;

  for (int kb = 0; kb < Kdim; kb += 128) {
    __builtin_amdgcn_s_wait_tensorcnt(0);   // own quarter of slab `cur` landed
    __syncthreads();                        // whole slab ready; other buf free
    if (kb + 128 < Kdim) {                  // kick next slab into other buffer
      tdm_load_2d((cur ? lds0 : lds1) + lds_off, gptr, flags, tile0, tile1, NV);
      gptr += gstep;
    }

#pragma unroll
    for (int t = 0; t < 4; ++t) {
      const int w0 = t * 32;
      // A fragment, 16x32, contiguous k order (same as B):
      //   half 0: elems 0..15 -> K w0+0..15
      //   half 1: elems 0..15 -> K w0+16..31
      v16bf a;
      if (!TRANSPOSED) {
        const float* lr = &adj_s[cur][ln16 * 129];
#pragma unroll
        for (int i = 0; i < 8; ++i) a[i] = (__bf16)lr[w0 + half * 16 + i];
#pragma unroll
        for (int i = 0; i < 8; ++i) a[8 + i] = (__bf16)lr[w0 + half * 16 + 8 + i];
      } else {
        const float* lc = &adj_s[cur][ln16];
#pragma unroll
        for (int i = 0; i < 8; ++i) a[i] = (__bf16)lc[(w0 + half * 16 + i) * 17];
#pragma unroll
        for (int i = 0; i < 8; ++i) a[8 + i] = (__bf16)lc[(w0 + half * 16 + 8 + i) * 17];
      }
      // B fragment, 32x16 layout: lane = column, halves split K 0..15 / 16..31
      v16bf bh = *(const v16bf*)(bh_row + kb + w0 + half * 16);
      acc = __builtin_amdgcn_wmma_f32_16x16x32_bf16(false, a, false, bh,
                                                    (short)0, acc, false, false);
      v16bf bl = *(const v16bf*)(bl_row + kb + w0 + half * 16);
      acc = __builtin_amdgcn_wmma_f32_16x16x32_bf16(false, a, false, bl,
                                                    (short)0, acc, false, false);
    }
    cur ^= 1;
  }

  // D layout: acc[j] = D[half*8 + j][ln16]
#pragma unroll
  for (int j = 0; j < 8; ++j) {
    int m = mbase + half * 8 + j;
    atomicAdd(&out[(size_t)m * DIM + wv * 16 + ln16], acc[j]);
  }
}

// ---------------------------------------------------------------------------
// logits[b][c] = z_u[u[b]] . Q[c] . z_v[v[b]];  one 64-thread block per pair
// ---------------------------------------------------------------------------
__global__ void decoder_kernel(const int* __restrict__ uidx,
                               const int* __restrict__ vidx,
                               const float* __restrict__ z_u,
                               const float* __restrict__ z_v,
                               const float* __restrict__ Q,
                               float* __restrict__ out) {
  __shared__ float zu_s[DIM], zv_s[DIM], red[NCLASS][DIM];
  const int b = blockIdx.x, d = threadIdx.x;
  zu_s[d] = z_u[(size_t)uidx[b] * DIM + d];
  zv_s[d] = z_v[(size_t)vidx[b] * DIM + d];
  __syncthreads();
  const float zu_d = zu_s[d];
#pragma unroll
  for (int c = 0; c < NCLASS; ++c) {
    const float* q = Q + ((size_t)c * DIM + d) * DIM;
    float s = 0.f;
#pragma unroll
    for (int e = 0; e < DIM; ++e) s += q[e] * zv_s[e];
    red[c][d] = zu_d * s;
  }
  __syncthreads();
  if (d < NCLASS) {
    float t = 0.f;
    for (int i = 0; i < DIM; ++i) t += red[d][i];
    out[(size_t)b * NCLASS + d] = t;
  }
}

// ---------------------------------------------------------------------------
extern "C" void kernel_launch(void* const* d_in, const int* in_sizes, int n_in,
                              void* d_out, int out_size, void* d_ws, size_t ws_size,
                              hipStream_t stream) {
  (void)in_sizes; (void)n_in; (void)out_size; (void)ws_size;

  const int*   u     = (const int*)d_in[0];
  const int*   v     = (const int*)d_in[1];
  const float* u_emb = (const float*)d_in[2];
  const float* v_emb = (const float*)d_in[3];
  const float* W1u   = (const float*)d_in[4];
  const float* b1u   = (const float*)d_in[5];
  const float* W1v   = (const float*)d_in[6];
  const float* b1v   = (const float*)d_in[7];
  const float* W2u   = (const float*)d_in[8];
  const float* b2u   = (const float*)d_in[9];
  const float* W2v   = (const float*)d_in[10];
  const float* b2v   = (const float*)d_in[11];
  const float* Q     = (const float*)d_in[12];
  const float* adj   = (const float*)d_in[13];
  float* out = (float*)d_out;

  char* ws = (char*)d_ws;
  const size_t msz = (size_t)NCLASS * DIM * NV * sizeof(__bf16); // 3.93 MB
  const size_t hsz = (size_t)NU * DIM * sizeof(float);           // 1.57 MB
  __bf16* mA_hi = (__bf16*)(ws);
  __bf16* mA_lo = (__bf16*)(ws + msz);
  __bf16* mB_hi = (__bf16*)(ws + 2 * msz);
  __bf16* mB_lo = (__bf16*)(ws + 3 * msz);
  float* h_u = (float*)(ws + 4 * msz);
  float* h_v = (float*)(ws + 4 * msz + hsz);
  float* z_u = (float*)(ws + 4 * msz + 2 * hsz);
  float* z_v = (float*)(ws + 4 * msz + 3 * hsz);

  const dim3 b64(64), b128(128), b256(256);
  const dim3 gElem((NU * DIM + 255) / 256);
  const dim3 gMsg(NU / 64, DIM, NCLASS);
  const dim3 gSpmm(NU / 16, NCLASS);

  // -------- layer 1 --------
  bias_init_kernel<<<gElem, b256, 0, stream>>>(b1v, h_u, NU * DIM);
  bias_init_kernel<<<gElem, b256, 0, stream>>>(b1u, h_v, NV * DIM);
  msgT_kernel<<<gMsg, b64, 0, stream>>>(v_emb, W1v, mA_hi, mA_lo, NV);
  msgT_kernel<<<gMsg, b64, 0, stream>>>(u_emb, W1u, mB_hi, mB_lo, NU);
  spmm_wmma_kernel<false><<<gSpmm, b128, 0, stream>>>(adj, mA_hi, mA_lo, h_u);
  spmm_wmma_kernel<true ><<<gSpmm, b128, 0, stream>>>(adj, mB_hi, mB_lo, h_v);
  relu_kernel<<<gElem, b256, 0, stream>>>(h_u, NU * DIM);
  relu_kernel<<<gElem, b256, 0, stream>>>(h_v, NV * DIM);

  // -------- layer 2 --------
  bias_init_kernel<<<gElem, b256, 0, stream>>>(b2u, z_u, NU * DIM);
  bias_init_kernel<<<gElem, b256, 0, stream>>>(b2v, z_v, NV * DIM);
  msgT_kernel<<<gMsg, b64, 0, stream>>>(h_v, W2u, mA_hi, mA_lo, NV);
  msgT_kernel<<<gMsg, b64, 0, stream>>>(h_u, W2v, mB_hi, mB_lo, NU);
  spmm_wmma_kernel<false><<<gSpmm, b128, 0, stream>>>(adj, mA_hi, mA_lo, z_u);
  spmm_wmma_kernel<true ><<<gSpmm, b128, 0, stream>>>(adj, mB_hi, mB_lo, z_v);

  // -------- bilinear decoder --------
  decoder_kernel<<<dim3(NB), b64, 0, stream>>>(u, v, z_u, z_v, Q, out);
}
